// Alignment_86157043958298
// MI455X (gfx1250) — matrix operand order repaired and checked by reading, not compiled, not run on hardware
//
#include <hip/hip_runtime.h>
#include <hip/hip_bf16.h>

// Problem sizes (fixed by the reference)
#define BB 32
#define SS 4096
#define EE 512
#define DD 512
#define HH 512

typedef __attribute__((ext_vector_type(16))) __bf16 v16bf;
typedef __attribute__((ext_vector_type(8)))  float  v8f;

// Types matching the async-to-LDS builtin signature:
//   arg0: int __attribute__((vector_size(16))) __device__ *   (AS1, global)
//   arg1: LDS-side pointer (AS3)
typedef int v4i __attribute__((vector_size(16)));
typedef __attribute__((address_space(1))) v4i glob_v4i;
typedef __attribute__((address_space(3))) v4i lds_v4i;

#if __has_builtin(__builtin_amdgcn_global_load_async_to_lds_b128)
#define USE_ASYNC_LDS 1
#endif

// ---------------- helpers ----------------
__device__ inline unsigned bf16_bits(float f) {
    unsigned x = __float_as_uint(f);
    unsigned r = x + 0x7FFFu + ((x >> 16) & 1u);   // RNE f32 -> bf16
    return r >> 16;
}
__device__ inline unsigned pack_bf16(float lo, float hi) {
    return bf16_bits(lo) | (bf16_bits(hi) << 16);
}
// A-fragment (16x32 MxK bf16) LDS index for the bf16 pair at even K=e within
// a 512-wide row; mm = row 0..15. Layout per ISA: VGPR j (j<4): K=2j+hf*8,
// (j>=4): K=16+2(j-4)+hf*8; lanes 0-15 = rows (hf=0), lanes 16-31 (hf=1).
__device__ inline int afrag_index(int e, int mm) {
    int kt = e >> 5;
    int q  = (e & 31) >> 1;
    int hf = (q >> 2) & 1;
    int j  = (q & 3) + ((q >> 3) << 2);
    return kt * 256 + (mm + hf * 16) * 8 + j;
}
__device__ inline lds_v4i* to_lds4(void* p) {
    // generic LDS pointer: low 32 bits are the LDS byte offset
    return (lds_v4i*)(unsigned)(size_t)p;
}
__device__ inline glob_v4i* to_glob4(const void* p) {
    return (glob_v4i*)(size_t)p;
}

// ---------------- dec_h = decoder_hidden @ W_dec^T + b_dec ----------------
__global__ __launch_bounds__(256) void dech_kernel(const float* __restrict__ dec,
                                                   const float* __restrict__ W_dec,
                                                   const float* __restrict__ b_dec,
                                                   float* __restrict__ dec_h) {
    int h = blockIdx.x * blockDim.x + threadIdx.x;   // grid.x = HH/256
    int b = blockIdx.y;
    const float4* wr = (const float4*)(W_dec + (size_t)h * DD);
    const float4* dr = (const float4*)(dec + (size_t)b * DD);
    float acc = 0.f;
#pragma unroll 4
    for (int d = 0; d < DD / 4; ++d) {
        float4 wv = wr[d], dv = dr[d];
        acc += wv.x * dv.x + wv.y * dv.y + wv.z * dv.z + wv.w * dv.w;
    }
    dec_h[b * HH + h] = acc + b_dec[h];
}

// ---------------- W_enc (f32, [H,E]) -> bf16 B-fragments ----------------
// wsB[(nt*16+kt)*256 + lane*8 + j]: B frag (32x16 KxN): lane n = column;
// VGPR j (j<4): K = half*8 + 2j, (j>=4): K = 16 + half*8 + 2(j-4).
__global__ __launch_bounds__(256) void swizzleW_kernel(const float* __restrict__ W_enc,
                                                       unsigned* __restrict__ wsB) {
    int idx = blockIdx.x * 256 + threadIdx.x;        // 0 .. 131071
    int j    = idx & 7;
    int lane = (idx >> 3) & 31;
    int kt   = (idx >> 8) & 15;
    int nt   = idx >> 12;
    int n = lane & 15, hf = lane >> 4;
    int h = nt * 16 + n;
    int e = kt * 32 + hf * 8 + 2 * (j & 3) + (j >> 2) * 16;
    wsB[idx] = pack_bf16(W_enc[(size_t)h * EE + e], W_enc[(size_t)h * EE + e + 1]);
}

// ---------------- fused projection + tanh + v-dot (WMMA) ----------------
// grid = (S/32, B), block = 256 (8 waves). Each block: 32 s-rows x H=512.
// Each wave owns 4 N-tiles and both M-tiles -> 8 WMMA per K-step, 128 total.
__global__ __launch_bounds__(256) void score_kernel(const float* __restrict__ enc,
                                                    const float* __restrict__ b_enc,
                                                    const float* __restrict__ v,
                                                    const float* __restrict__ dec_h,
                                                    const unsigned* __restrict__ wsB,
                                                    float* __restrict__ scores,
                                                    int* __restrict__ flags) {
#ifdef USE_ASYNC_LDS
    __shared__ alignas(16) float ldsRaw[32 * EE];     // 64 KB raw f32 tile
#endif
    __shared__ alignas(32) unsigned ldsA[2][16 * 256]; // 2 x 16KB bf16 A-fragments
    __shared__ float lds_sc[32];
    __shared__ int   lds_fl[32];

    int tid = threadIdx.x;
    int s0  = blockIdx.x * 32;
    int b   = blockIdx.y;
    if (tid < 32) { lds_sc[tid] = 0.f; lds_fl[tid] = 0; }

    int lane = tid & 31;
    int wv   = tid >> 5;           // wave id 0..7
    int ln   = lane & 15;
    int hf   = lane >> 4;

    const float* encT = enc + ((size_t)b * SS + s0) * EE;   // 32 x 512 tile

#ifdef USE_ASYNC_LDS
    // ---- coalesced async copy: global -> LDS (raw, row-major) ----
#pragma unroll
    for (int i = 0; i < 16; ++i) {
        int f  = tid + 256 * i;            // float4 index, 0..4095
        int m  = f >> 7;                   // row 0..31 (128 float4 per row)
        int f4 = f & 127;
        __builtin_amdgcn_global_load_async_to_lds_b128(
            to_glob4(encT + (size_t)m * EE + f4 * 4),
            to_lds4(&ldsRaw[m * EE + f4 * 4]), 0, 0);
    }
#if __has_builtin(__builtin_amdgcn_s_wait_asynccnt)
    __builtin_amdgcn_s_wait_asynccnt(0);
#else
    asm volatile("s_wait_asynccnt 0" ::: "memory");
#endif
    __syncthreads();
    // ---- LDS raw f32 -> LDS bf16 A-fragments ----
#pragma unroll
    for (int i = 0; i < 32; ++i) {
        int p = tid + 256 * i;             // bf16-pair index, 0..8191
        int m = p >> 8;                    // row 0..31 (256 pairs per row)
        int e = (p & 255) * 2;
        float lo = ldsRaw[m * EE + e], hi = ldsRaw[m * EE + e + 1];
        ldsA[m >> 4][afrag_index(e, m & 15)] = pack_bf16(lo, hi);
    }
#else
    // ---- fallback: coalesced b128 register staging, direct fragment store ----
#pragma unroll
    for (int i = 0; i < 16; ++i) {
        int f  = tid + 256 * i;
        int m  = f >> 7;
        int f4 = f & 127;
        float4 d = *(const float4*)(encT + (size_t)m * EE + f4 * 4);
        int e = f4 * 4;
        ldsA[m >> 4][afrag_index(e,     m & 15)] = pack_bf16(d.x, d.y);
        ldsA[m >> 4][afrag_index(e + 2, m & 15)] = pack_bf16(d.z, d.w);
    }
#endif
    __syncthreads();

    // ---- WMMA main loop ----
    int ntBase = wv * 4;
    const unsigned* Bb = wsB + (size_t)ntBase * 4096 + lane * 8;
    v8f acc0[4] = {}, acc1[4] = {};
    for (int kt = 0; kt < 16; ++kt) {
        v16bf a0 = *(const v16bf*)&ldsA[0][kt * 256 + lane * 8];
        v16bf a1 = *(const v16bf*)&ldsA[1][kt * 256 + lane * 8];
        const unsigned* bp = Bb + kt * 256;
        __builtin_prefetch(bp + 256, 0, 1);           // global_prefetch_b8
#pragma unroll
        for (int t = 0; t < 4; ++t) {
            v16bf bf = *(const v16bf*)(bp + t * 4096);
            acc0[t] = __builtin_amdgcn_wmma_f32_16x16x32_bf16(
                false, a0, false, bf, (short)0, acc0[t], false, false);
            acc1[t] = __builtin_amdgcn_wmma_f32_16x16x32_bf16(
                false, a1, false, bf, (short)0, acc1[t], false, false);
        }
    }

    // ---- epilogue: +b_enc, nonzero flag, tanh(+dec_h)*v, reduce over h ----
    float part0[8], part1[8];
    int   fl0[8], fl1[8];
#pragma unroll
    for (int r = 0; r < 8; ++r) { part0[r] = part1[r] = 0.f; fl0[r] = fl1[r] = 0; }

#pragma unroll
    for (int t = 0; t < 4; ++t) {
        int h = (ntBase + t) * 16 + ln;
        float be = b_enc[h];
        float dh = dec_h[b * HH + h];
        float vv = v[h];
#pragma unroll
        for (int r = 0; r < 8; ++r) {
            float e0 = acc0[t][r] + be;
            float e1 = acc1[t][r] + be;
            fl0[r] |= (e0 != 0.0f) ? 1 : 0;
            fl1[r] |= (e1 != 0.0f) ? 1 : 0;
            part0[r] += tanhf(e0 + dh) * vv;
            part1[r] += tanhf(e1 + dh) * vv;
        }
    }
#pragma unroll
    for (int r = 0; r < 8; ++r) {
        float v0 = part0[r], v1 = part1[r];
        int f0 = fl0[r], f1 = fl1[r];
        for (int off = 1; off < 16; off <<= 1) {
            v0 += __shfl_xor(v0, off, 32);
            v1 += __shfl_xor(v1, off, 32);
            f0 |= __shfl_xor(f0, off, 32);
            f1 |= __shfl_xor(f1, off, 32);
        }
        if (ln == 0) {                                 // lanes 0 and 16
            int row = r + 8 * hf;
            atomicAdd(&lds_sc[row], v0);
            atomicOr(&lds_fl[row], f0);
            atomicAdd(&lds_sc[16 + row], v1);
            atomicOr(&lds_fl[16 + row], f1);
        }
    }
    __syncthreads();
    if (tid < 32) {
        size_t o = (size_t)b * SS + s0 + tid;
        scores[o] = lds_sc[tid];
        flags[o]  = lds_fl[tid];
    }
}

// ---------------- ragged softmax over S (cumprod-valid semantics) ----------------
__global__ __launch_bounds__(256) void softmax_kernel(float* __restrict__ scores,
                                                      const int* __restrict__ flags) {
    __shared__ float red[256];
    __shared__ int   redi[256];
    int b = blockIdx.x, tid = threadIdx.x;
    float*     sc = scores + (size_t)b * SS;
    const int* fl = flags + (size_t)b * SS;

    int lmin = SS;
    for (int s = tid; s < SS; s += 256)
        if (fl[s] == 0) lmin = min(lmin, s);
    redi[tid] = lmin; __syncthreads();
    for (int st = 128; st > 0; st >>= 1) {
        if (tid < st) redi[tid] = min(redi[tid], redi[tid + st]);
        __syncthreads();
    }
    int L = redi[0];

    float mx = -__builtin_inff();
    for (int s = tid; s < L; s += 256) mx = fmaxf(mx, sc[s]);
    red[tid] = mx; __syncthreads();
    for (int st = 128; st > 0; st >>= 1) {
        if (tid < st) red[tid] = fmaxf(red[tid], red[tid + st]);
        __syncthreads();
    }
    mx = red[0]; __syncthreads();

    float sum = 0.f;
    for (int s = tid; s < L; s += 256) sum += __expf(sc[s] - mx);
    red[tid] = sum; __syncthreads();
    for (int st = 128; st > 0; st >>= 1) {
        if (tid < st) red[tid] += red[tid + st];
        __syncthreads();
    }
    float inv = (red[0] > 0.f) ? 1.0f / red[0] : 0.f;
    __syncthreads();

    for (int s = tid; s < SS; s += 256)
        sc[s] = (s < L) ? __expf(sc[s] - mx) * inv : 0.f;   // attn (in place)
}

// ---------------- context = sum_s attn[s] * enc[b,s,:] ----------------
__global__ void zero_kernel(float* __restrict__ out, int n) {
    int i = blockIdx.x * 256 + threadIdx.x;
    if (i < n) out[i] = 0.f;
}

__global__ __launch_bounds__(128) void context_kernel(const float* __restrict__ enc,
                                                      const float* __restrict__ attn,
                                                      float* __restrict__ out) {
    int e  = blockIdx.x * 128 + threadIdx.x;   // grid.x = E/128
    int sc = blockIdx.y;                       // grid.y = S/512 chunks
    int b  = blockIdx.z;
    const float* encB = enc + ((size_t)b * SS + (size_t)sc * 512) * EE;
    const float* at   = attn + (size_t)b * SS + sc * 512;
    float acc = 0.f;
#pragma unroll 4
    for (int s = 0; s < 512; ++s)
        acc += at[s] * encB[(size_t)s * EE + e];
    atomicAdd(&out[b * EE + e], acc);
}

// ---------------- launcher ----------------
extern "C" void kernel_launch(void* const* d_in, const int* in_sizes, int n_in,
                              void* d_out, int out_size, void* d_ws, size_t ws_size,
                              hipStream_t stream) {
    const float* enc   = (const float*)d_in[0];   // [B,S,E]
    const float* dec   = (const float*)d_in[1];   // [B,D]
    const float* W_enc = (const float*)d_in[2];   // [H,E]
    const float* b_enc = (const float*)d_in[3];   // [H]
    const float* W_dec = (const float*)d_in[4];   // [H,D]
    const float* b_dec = (const float*)d_in[5];   // [H]
    const float* v     = (const float*)d_in[6];   // [H]
    float* out = (float*)d_out;                   // [B,1,E]

    char* w = (char*)d_ws;
    float*    dec_h  = (float*)w;                               // 64 KB
    unsigned* wsB    = (unsigned*)(w + (64 << 10));             // 512 KB bf16 fragments
    float*    scores = (float*)(w + (64 << 10) + (512 << 10));  // 512 KB
    int*      flags  = (int*)(w + (64 << 10) + (1024 << 10));   // 512 KB

    zero_kernel<<<dim3((BB * EE + 255) / 256), 256, 0, stream>>>(out, BB * EE);
    dech_kernel<<<dim3(HH / 256, BB), 256, 0, stream>>>(dec, W_dec, b_dec, dec_h);
    swizzleW_kernel<<<dim3((HH / 16) * (EE / 32)), 256, 0, stream>>>(W_enc, wsB);
    score_kernel<<<dim3(SS / 32, BB), 256, 0, stream>>>(enc, b_enc, v, dec_h, wsB,
                                                        scores, flags);
    softmax_kernel<<<dim3(BB), 256, 0, stream>>>(scores, flags);
    context_kernel<<<dim3(EE / 128, SS / 512, BB), 128, 0, stream>>>(enc, scores, out);
}
